// GAT_30039001268364
// MI455X (gfx1250) — compile-verified, matrix operand order checked by reading, not run on hardware
//
#include <hip/hip_runtime.h>
#include <hip/hip_bf16.h>

// ---------------- problem constants (from reference) ----------------
constexpr int NN    = 50000;   // nodes
constexpr int EE    = 800000;  // edges (before self loops)
constexpr int ETOT  = EE + NN; // edges incl. self loops
constexpr int HEADS = 8;
constexpr int HID   = 128;
constexpr int IN_D  = 128;
constexpr int OUT_D = 10;
constexpr int GRAPHS= 64;
#define BN_EPS  1e-5f
#define SLOPE   0.2f

typedef __attribute__((ext_vector_type(16))) _Float16 v16h;
typedef __attribute__((ext_vector_type(8)))  float    v8f;

// ---------------- small helpers ----------------
__device__ __forceinline__ float atomicAddF(float* p, float v) {
  // agent-scope relaxed fadd -> global_atomic_add_f32 on gfx1250
  return __hip_atomic_fetch_add(p, v, __ATOMIC_RELAXED, __HIP_MEMORY_SCOPE_AGENT);
}
__device__ __forceinline__ void atomicMaxF(float* addr, float val) {
  // classic signed/unsigned bit trick for float max
  if (val >= 0.0f) atomicMax((int*)addr, __float_as_int(val));
  else             atomicMin((unsigned int*)addr, __float_as_uint(val));
}
__device__ __forceinline__ float leaky(float v) { return v > 0.0f ? v : SLOPE * v; }

// ---------------- generic utility kernels ----------------
__global__ void k_fill_u32(unsigned int* p, unsigned int pat, int n) {
  int i = blockIdx.x * blockDim.x + threadIdx.x;
  if (i < n) p[i] = pat;
}
__global__ void k_f32_to_f16(const float* __restrict__ s, _Float16* __restrict__ d, int n) {
  int i = blockIdx.x * blockDim.x + threadIdx.x;
  if (i < n) d[i] = (_Float16)s[i];
}

// Pack a KxNcols f32 weight into WMMA B-fragment layout:
// Bp[((nt*KC + kc)*32 + lane)*16 + e]   (per ISA 16-bit B layout, wave32)
__global__ void k_pack_b(const float* __restrict__ B, _Float16* __restrict__ Bp,
                         int K, int Ncols) {
  int KC = K >> 5, NT = Ncols >> 4;
  int total = NT * KC * 32 * 16;
  int idx = blockIdx.x * blockDim.x + threadIdx.x;
  if (idx >= total) return;
  int e    = idx & 15;
  int lane = (idx >> 4) & 31;
  int rest = idx >> 9;
  int kc   = rest % KC;
  int nt   = rest / KC;
  int n     = (nt << 4) + (lane & 15);
  int khalf = (lane >> 4) << 3;                       // 0 or 8
  int k     = (kc << 5) + khalf + ((e >> 3) << 4) + (e & 7);
  Bp[idx] = (_Float16)B[k * Ncols + n];
}

// ---------------- WMMA GEMM: C[Mx128] = A_f16[MxK] * Bpacked + bias ----------------
// Each wave computes a full 16x128 output strip: one A fragment per K-chunk is
// reused across all 8 n-tiles (8 back-to-back WMMAs), cutting A traffic 8x.
// 4 waves per block, one m-tile per wave. M%16==0, Ncols==128, K%32==0.
__global__ void k_wmma_gemm(const _Float16* __restrict__ A,
                            const _Float16* __restrict__ Bp,
                            const float* __restrict__ bias,
                            float* __restrict__ C,
                            int M, int Ncols, int K, int relu) {
  constexpr int NT = 8;                  // Ncols / 16 == 8 for all our GEMMs
  int wave = threadIdx.x >> 5;
  int lane = threadIdx.x & 31;
  int mt = blockIdx.x * 4 + wave;        // wave-uniform
  if (mt >= (M >> 4)) return;
  int m0 = mt << 4;
  int row   = m0 + (lane & 15);
  int khalf = (lane >> 4) << 3;          // 0 or 8
  int KC = K >> 5;
  v8f zero = {};
  v8f acc[NT];
#pragma unroll
  for (int t = 0; t < NT; ++t) acc[t] = zero;
  for (int kc = 0; kc < KC; ++kc) {
    int kbase = (kc << 5) + khalf;
    const _Float16* ap = A + (size_t)row * K + kbase;
    v16h a;
#pragma unroll
    for (int j = 0; j < 8; ++j) { a[j] = ap[j]; a[j + 8] = ap[16 + j]; }
#pragma unroll
    for (int t = 0; t < NT; ++t) {
      const _Float16* bp = Bp + (((size_t)(t * KC + kc) * 32) + lane) * 16;
      v16h b;
#pragma unroll
      for (int j = 0; j < 16; ++j) b[j] = bp[j];
      acc[t] = __builtin_amdgcn_wmma_f32_16x16x32_f16(false, a, false, b,
                                                      (short)0, acc[t], false, false);
    }
  }
  int rbase = m0 + ((lane >> 4) << 3);   // lanes>=16 hold rows m0+8..m0+15
#pragma unroll
  for (int t = 0; t < NT; ++t) {
    int col = (t << 4) + (lane & 15);
    float bv = bias ? bias[col] : 0.0f;
#pragma unroll
    for (int r = 0; r < 8; ++r) {
      float v = acc[t][r] + bv;
      if (relu) v = fmaxf(v, 0.0f);
      C[(size_t)(rbase + r) * Ncols + col] = v;
    }
  }
}

// ---------------- GAT pieces ----------------
// per (node, head) attention logits
__global__ void k_att_scores(const float* __restrict__ t,
                             const float* __restrict__ attS,
                             const float* __restrict__ attD,
                             float* __restrict__ asrc, float* __restrict__ adst,
                             int heads, int ch) {
  int idx = blockIdx.x * blockDim.x + threadIdx.x;
  if (idx >= NN * heads) return;
  int n = idx / heads, h0 = idx - n * heads;
  const float* tp = t + (size_t)n * HID + h0 * ch;
  float ss = 0.f, dd = 0.f;
  for (int c = 0; c < ch; ++c) {
    float v = tp[c];
    ss += v * attS[h0 * ch + c];
    dd += v * attD[h0 * ch + c];
  }
  asrc[idx] = ss; adst[idx] = dd;
}

__device__ __forceinline__ void edge_sd(const int* __restrict__ ei, int e, int& s, int& d) {
  if (e < EE) { s = ei[e]; d = ei[EE + e]; } else { s = d = e - EE; }
}

__global__ void k_edge_max(const int* __restrict__ ei,
                           const float* __restrict__ asrc, const float* __restrict__ adst,
                           float* __restrict__ amax, int heads) {
  int idx = blockIdx.x * blockDim.x + threadIdx.x;
  if (idx >= ETOT * heads) return;
  int e = idx / heads, h0 = idx - e * heads;
  int s, d; edge_sd(ei, e, s, d);
  float v = leaky(asrc[s * heads + h0] + adst[d * heads + h0]);
  atomicMaxF(&amax[d * heads + h0], v);
}

__global__ void k_edge_sum(const int* __restrict__ ei,
                           const float* __restrict__ asrc, const float* __restrict__ adst,
                           const float* __restrict__ amax, float* __restrict__ denom,
                           int heads) {
  int idx = blockIdx.x * blockDim.x + threadIdx.x;
  if (idx >= ETOT * heads) return;
  int e = idx / heads, h0 = idx - e * heads;
  int s, d; edge_sd(ei, e, s, d);
  float v = leaky(asrc[s * heads + h0] + adst[d * heads + h0]);
  atomicAddF(&denom[d * heads + h0], __expf(v - amax[d * heads + h0]));
}

__global__ void k_edge_scatter(const int* __restrict__ ei,
                               const float* __restrict__ asrc, const float* __restrict__ adst,
                               const float* __restrict__ amax, const float* __restrict__ denom,
                               const float* __restrict__ t, float* __restrict__ cout,
                               int heads, int ch) {
  int idx = blockIdx.x * blockDim.x + threadIdx.x;
  if (idx >= ETOT * heads) return;
  int e = idx / heads, h0 = idx - e * heads;
  int s, d; edge_sd(ei, e, s, d);
  float v = leaky(asrc[s * heads + h0] + adst[d * heads + h0]);
  float alpha = __expf(v - amax[d * heads + h0]) / denom[d * heads + h0];
  const float* tp = t + (size_t)s * HID + h0 * ch;
  float* op = cout + (size_t)d * HID + h0 * ch;
  for (int c = 0; c < ch; ++c) atomicAddF(&op[c], tp[c] * alpha);
}

// BatchNorm stats: one block per channel (128 blocks), LDS tree reduction
__global__ void k_bn_stats(const float* __restrict__ x, float* __restrict__ mean,
                           float* __restrict__ inv) {
  __shared__ float s1[256], s2[256];
  int j = blockIdx.x;
  float a = 0.f, b = 0.f;
  for (int n = threadIdx.x; n < NN; n += 256) {
    float v = x[(size_t)n * HID + j];
    a += v; b += v * v;
  }
  s1[threadIdx.x] = a; s2[threadIdx.x] = b;
  __syncthreads();
  for (int s = 128; s > 0; s >>= 1) {
    if (threadIdx.x < s) { s1[threadIdx.x] += s1[threadIdx.x + s];
                           s2[threadIdx.x] += s2[threadIdx.x + s]; }
    __syncthreads();
  }
  if (threadIdx.x == 0) {
    float m = s1[0] / (float)NN;
    float v = s2[0] / (float)NN - m * m;     // biased var, matches jnp.var
    mean[j] = m;
    inv[j]  = rsqrtf(v + BN_EPS);
  }
}

// fused BN apply + ReLU + residual; also emits f16 copy of new h for next GEMM
__global__ void k_bn_apply(const float* __restrict__ c, const float* __restrict__ mean,
                           const float* __restrict__ inv, const float* __restrict__ gamma,
                           const float* __restrict__ beta,
                           float* __restrict__ h, _Float16* __restrict__ h16) {
  int idx = blockIdx.x * blockDim.x + threadIdx.x;
  if (idx >= NN * HID) return;
  int j = idx & (HID - 1);
  float y = (c[idx] - mean[j]) * inv[j] * gamma[j] + beta[j];
  y = fmaxf(y, 0.0f);
  float hn = y + h[idx];
  h[idx]  = hn;
  h16[idx] = (_Float16)hn;
}

// ---------------- pooling + head ----------------
__global__ void k_pool_cnt(const int* __restrict__ batch, float* __restrict__ cnt) {
  int n = blockIdx.x * blockDim.x + threadIdx.x;
  if (n < NN) atomicAddF(&cnt[batch[n]], 1.0f);
}
__global__ void k_pool_sum(const int* __restrict__ batch, const float* __restrict__ h,
                           float* __restrict__ pooled) {
  int idx = blockIdx.x * blockDim.x + threadIdx.x;
  if (idx >= NN * HID) return;
  int n = idx >> 7, j = idx & (HID - 1);
  atomicAddF(&pooled[(size_t)batch[n] * HID + j], h[idx]);
}
__global__ void k_pool_div(float* __restrict__ pooled, const float* __restrict__ cnt,
                           _Float16* __restrict__ pooled16) {
  int idx = blockIdx.x * blockDim.x + threadIdx.x;
  if (idx >= GRAPHS * HID) return;
  int g = idx >> 7;
  float p = pooled[idx] / fmaxf(cnt[g], 1.0f);
  pooled[idx] = p;
  pooled16[idx] = (_Float16)p;
}
__global__ void k_lin3(const float* __restrict__ x, const float* __restrict__ W,
                       const float* __restrict__ b, float* __restrict__ out) {
  int idx = blockIdx.x * blockDim.x + threadIdx.x;
  if (idx >= GRAPHS * OUT_D) return;
  int g = idx / OUT_D, o = idx - g * OUT_D;
  float acc = b[o];
  for (int k = 0; k < HID; ++k) acc += x[g * HID + k] * W[k * OUT_D + o];
  out[idx] = acc;
}

// ---------------- host side ----------------
static inline int cdiv(int a, int b) { return (a + b - 1) / b; }

extern "C" void kernel_launch(void* const* d_in, const int* in_sizes, int n_in,
                              void* d_out, int out_size, void* d_ws, size_t ws_size,
                              hipStream_t stream) {
  (void)in_sizes; (void)n_in; (void)out_size; (void)ws_size;
  const float* x        = (const float*)d_in[0];
  const int*   ei       = (const int*)  d_in[1];
  const int*   batch    = (const int*)  d_in[3];
  const float* W_emb    = (const float*)d_in[4];
  const float* b_emb    = (const float*)d_in[5];
  const float* conv_W   = (const float*)d_in[6];
  const float* c_attS   = (const float*)d_in[7];
  const float* c_attD   = (const float*)d_in[8];
  const float* conv2_W  = (const float*)d_in[10];
  const float* c2_attS  = (const float*)d_in[11];
  const float* c2_attD  = (const float*)d_in[12];
  const float* bn_gamma = (const float*)d_in[14];
  const float* bn_beta  = (const float*)d_in[15];
  const float* lin1_W   = (const float*)d_in[16];
  const float* lin1_b   = (const float*)d_in[17];
  const float* lin2_W   = (const float*)d_in[18];
  const float* lin2_b   = (const float*)d_in[19];
  const float* lin3_W   = (const float*)d_in[20];
  const float* lin3_b   = (const float*)d_in[21];
  float* out = (float*)d_out;

  // ---- workspace carve ----
  char* base = (char*)d_ws; size_t off = 0;
  auto carve = [&](size_t bytes) { void* p = base + off; off = (off + bytes + 255) & ~(size_t)255; return p; };
  float*    h_f32   = (float*)   carve((size_t)NN * HID * 4);
  float*    t_f32   = (float*)   carve((size_t)NN * HID * 4);
  float*    cout_f  = (float*)   carve((size_t)NN * HID * 4);
  _Float16* h_f16   = (_Float16*)carve((size_t)NN * HID * 2);
  float*    asrc    = (float*)   carve((size_t)NN * HEADS * 4);
  float*    adst    = (float*)   carve((size_t)NN * HEADS * 4);
  float*    amax    = (float*)   carve((size_t)NN * HEADS * 4);
  float*    denom   = (float*)   carve((size_t)NN * HEADS * 4);
  float*    mean    = (float*)   carve(HID * 4);
  float*    invs    = (float*)   carve(HID * 4);
  const size_t WPK  = (size_t)HID * HID * 2;   // 128x128 f16 packed
  _Float16* p_emb   = (_Float16*)carve(WPK);
  _Float16* p_c0    = (_Float16*)carve(WPK);
  _Float16* p_c1    = (_Float16*)carve(WPK);
  _Float16* p_c2    = (_Float16*)carve(WPK);
  _Float16* p_l1    = (_Float16*)carve(WPK);
  _Float16* p_l2    = (_Float16*)carve(WPK);
  float*    pooled  = (float*)   carve((size_t)GRAPHS * HID * 4);
  float*    cnt     = (float*)   carve(GRAPHS * 4);
  _Float16* pooled16= (_Float16*)carve((size_t)GRAPHS * HID * 2);
  float*    mlp1    = (float*)   carve((size_t)GRAPHS * HID * 4);
  _Float16* mlp1_16 = (_Float16*)carve((size_t)GRAPHS * HID * 2);
  float*    mlp2    = (float*)   carve((size_t)GRAPHS * HID * 4);

  const int TB = 256;
  const int packTot = (HID / 16) * (HID / 32) * 32 * 16;

  // ---- pack weights into WMMA B-fragment layout ----
  k_pack_b<<<cdiv(packTot, TB), TB, 0, stream>>>(W_emb,   p_emb, IN_D, HID);
  k_pack_b<<<cdiv(packTot, TB), TB, 0, stream>>>(conv_W,              p_c0, HID, HID);
  k_pack_b<<<cdiv(packTot, TB), TB, 0, stream>>>(conv_W + HID * HID,  p_c1, HID, HID);
  k_pack_b<<<cdiv(packTot, TB), TB, 0, stream>>>(conv2_W, p_c2, HID, HID);
  k_pack_b<<<cdiv(packTot, TB), TB, 0, stream>>>(lin1_W,  p_l1, HID, HID);
  k_pack_b<<<cdiv(packTot, TB), TB, 0, stream>>>(lin2_W,  p_l2, HID, HID);

  const int gemmBlocksN = cdiv(NN / 16, 4);     // 4 m-tiles (waves) per block
  const int gemmBlocksG = cdiv(GRAPHS / 16, 4); // MLP: 4 m-tiles in one block

  // ---- embedding: h = x @ W_emb + b_emb ----
  k_f32_to_f16<<<cdiv(NN * IN_D, TB), TB, 0, stream>>>(x, h_f16, NN * IN_D);
  k_wmma_gemm<<<gemmBlocksN, 128, 0, stream>>>(h_f16, p_emb, b_emb, h_f32,
                                               NN, HID, IN_D, 0);
  k_f32_to_f16<<<cdiv(NN * HID, TB), TB, 0, stream>>>(h_f32, h_f16, NN * HID);

  // ---- 3 GAT layers ----
  for (int layer = 0; layer < 3; ++layer) {
    int heads = (layer < 2) ? HEADS : 1;
    int ch    = HID / heads;
    const _Float16* Wp = (layer == 0) ? p_c0 : (layer == 1) ? p_c1 : p_c2;
    const float* aS = (layer < 2) ? (c_attS + layer * HID) : c2_attS;
    const float* aD = (layer < 2) ? (c_attD + layer * HID) : c2_attD;

    k_wmma_gemm<<<gemmBlocksN, 128, 0, stream>>>(h_f16, Wp, nullptr, t_f32,
                                                 NN, HID, HID, 0);

    k_att_scores<<<cdiv(NN * heads, TB), TB, 0, stream>>>(t_f32, aS, aD, asrc, adst, heads, ch);

    k_fill_u32<<<cdiv(NN * heads, TB), TB, 0, stream>>>((unsigned*)amax, 0xFF800000u, NN * heads);
    hipMemsetAsync(denom, 0, (size_t)NN * heads * 4, stream);
    hipMemsetAsync(cout_f, 0, (size_t)NN * HID * 4, stream);

    int ethreads = ETOT * heads;
    k_edge_max    <<<cdiv(ethreads, TB), TB, 0, stream>>>(ei, asrc, adst, amax, heads);
    k_edge_sum    <<<cdiv(ethreads, TB), TB, 0, stream>>>(ei, asrc, adst, amax, denom, heads);
    k_edge_scatter<<<cdiv(ethreads, TB), TB, 0, stream>>>(ei, asrc, adst, amax, denom,
                                                          t_f32, cout_f, heads, ch);

    k_bn_stats<<<HID, 256, 0, stream>>>(cout_f, mean, invs);
    k_bn_apply<<<cdiv(NN * HID, TB), TB, 0, stream>>>(cout_f, mean, invs,
                                                      bn_gamma + layer * HID,
                                                      bn_beta  + layer * HID,
                                                      h_f32, h_f16);
  }

  // ---- global mean pool ----
  hipMemsetAsync(pooled, 0, (size_t)GRAPHS * HID * 4, stream);
  hipMemsetAsync(cnt,    0, GRAPHS * 4, stream);
  k_pool_cnt<<<cdiv(NN, TB), TB, 0, stream>>>(batch, cnt);
  k_pool_sum<<<cdiv(NN * HID, TB), TB, 0, stream>>>(batch, h_f32, pooled);
  k_pool_div<<<cdiv(GRAPHS * HID, TB), TB, 0, stream>>>(pooled, cnt, pooled16);

  // ---- MLP head ----
  k_wmma_gemm<<<gemmBlocksG, 128, 0, stream>>>(pooled16, p_l1, lin1_b, mlp1,
                                               GRAPHS, HID, HID, 1);
  k_f32_to_f16<<<cdiv(GRAPHS * HID, TB), TB, 0, stream>>>(mlp1, mlp1_16, GRAPHS * HID);
  k_wmma_gemm<<<gemmBlocksG, 128, 0, stream>>>(mlp1_16, p_l2, lin2_b, mlp2,
                                               GRAPHS, HID, HID, 1);
  k_lin3<<<cdiv(GRAPHS * OUT_D, TB), TB, 0, stream>>>(mlp2, lin3_W, lin3_b, out);
}
